// LSTMModel_76441827934813
// MI455X (gfx1250) — compile-verified, matrix operand order checked by reading, not run on hardware
//
#include <hip/hip_runtime.h>
#include <hip/hip_bf16.h>

// CDNA5 / gfx1250 LSTM: 2 layers, B=1024, T=256, I=14, H=128, C=1.
// One workgroup per 16 batch rows; full time loop inside the block.
// Matmuls: v_wmma_f32_16x16x32_bf16; weights + h resident in LDS;
// next-timestep input tile prefetched (async global->LDS on layer 1).

typedef __attribute__((ext_vector_type(16))) __bf16 v16bf;
typedef __attribute__((ext_vector_type(8)))  float  v8f;
typedef __attribute__((ext_vector_type(4)))  int    v4i;

#define B_SZ 1024
#define T_SZ 256
#define I_SZ 14
#define H_SZ 128
#define G4   512   // 4*H (gate rows)

#define AS1 __attribute__((address_space(1)))
#define AS3 __attribute__((address_space(3)))

#if defined(__has_builtin)
#if __has_builtin(__builtin_amdgcn_global_load_async_to_lds_b128)
#define HAVE_ASYNC_LDS 1
#endif
#endif

__device__ __forceinline__ unsigned short f2bf(float f) {
    union { float f; unsigned u; } v; v.f = f;
    unsigned r = v.u + 0x7FFFu + ((v.u >> 16) & 1u);   // round-to-nearest-even
    return (unsigned short)(r >> 16);
}
__device__ __forceinline__ float bf2f(unsigned short b) {
    union { unsigned u; float f; } v; v.u = ((unsigned)b) << 16;
    return v.f;
}

union Frag { uint4 q[2]; v16bf v; };

// One 16-lane-striped bf16 WMMA fragment slice (16 elems = two 16B chunks).
__device__ __forceinline__ v16bf load_frag(const unsigned short* p) {
    Frag f;
    f.q[0] = *reinterpret_cast<const uint4*>(p);
    f.q[1] = *reinterpret_cast<const uint4*>(p + 16);
    return f.v;
}

__device__ __forceinline__ float sigmoidf_(float x) { return 1.0f / (1.0f + __expf(-x)); }
__device__ __forceinline__ float tanhf_(float x)    { return 2.0f / (1.0f + __expf(-2.0f * x)) - 1.0f; }

#ifdef HAVE_ASYNC_LDS
__device__ __forceinline__ void async_copy16(const unsigned short* g, unsigned short* l) {
    // builtin expects (AS1 v4i* src, AS3 v4i* dst, imm offset, imm cpol)
    __builtin_amdgcn_global_load_async_to_lds_b128((AS1 v4i*)g, (AS3 v4i*)l, 0, 0);
}
__device__ __forceinline__ void wait_async0() {
#if __has_builtin(__builtin_amdgcn_s_wait_asynccnt)
    __builtin_amdgcn_s_wait_asynccnt(0);
#else
    asm volatile("s_wait_asynccnt 0x0" ::: "memory");
#endif
}
#endif

// LAYER==0: input = x [B][T][14] f32 (K padded to 32), writes h1 seq (bf16) to hout.
// LAYER==1: input = h1 [B][T][128] bf16 from ws; also computes final linear -> out.
template <int LAYER>
__global__ __launch_bounds__(256)
void lstm_layer_kernel(const float* __restrict__ xin,
                       const unsigned short* __restrict__ hin,
                       const float* __restrict__ w_ih,
                       const float* __restrict__ w_hh,
                       const float* __restrict__ b_ih,
                       const float* __restrict__ b_hh,
                       unsigned short* __restrict__ hout,
                       const float* __restrict__ w_lin,
                       const float* __restrict__ b_lin,
                       float* __restrict__ out) {
    constexpr int KIN  = (LAYER == 0) ? 32 : H_SZ;     // input-proj K (padded for L0)
    constexpr int TILE = 16 * KIN;                     // one input tile (elements)

    extern __shared__ __align__(16) unsigned short smem[];
    unsigned short* s_wih = smem;                       // [G4][KIN]
    unsigned short* s_whh = s_wih + G4 * KIN;           // [G4][128]
    unsigned short* s_h   = s_whh + G4 * H_SZ;          // [16][128] recurrent h (bf16)
    unsigned short* s_in  = s_h   + 16 * H_SZ;          // 2 x TILE double buffer

    const int tid  = threadIdx.x;
    const int wave = tid >> 5;          // 0..7 : owns h-cols [16*wave, 16*wave+16)
    const int lane = tid & 31;
    const int col  = lane & 15;         // N (for D/B) and M (for A) index
    const int half = lane >> 4;         // fragment half selector
    const int row0 = blockIdx.x * 16;   // batch-row base of this block

    // layer-0 staging thread mapping (224 active threads)
    const int xm = tid / I_SZ, xk = tid - xm * I_SZ;
    // layer-1 staging thread mapping (256 threads x 16B)
    const int hm = tid >> 4, hc8 = (tid & 15) * 8;

#ifdef HAVE_ASYNC_LDS
    if (LAYER == 1) {   // kick off tile 0 before weight staging to hide latency
        async_copy16(hin + ((size_t)(row0 + hm) * T_SZ + 0) * H_SZ + hc8,
                     s_in + hm * H_SZ + hc8);
    }
#endif

    // ---- stage weights into LDS (bf16), zero h / input pads ----
    for (int idx = tid; idx < G4 * KIN; idx += 256) {
        int n = idx / KIN, k = idx % KIN;
        float v = (LAYER == 0) ? ((k < I_SZ) ? w_ih[n * I_SZ + k] : 0.0f)
                               : w_ih[n * H_SZ + k];
        s_wih[idx] = f2bf(v);
    }
    for (int idx = tid; idx < G4 * H_SZ; idx += 256) s_whh[idx] = f2bf(w_hh[idx]);
    for (int idx = tid; idx < 16 * H_SZ; idx += 256) s_h[idx] = 0;
    if (LAYER == 0) {   // zero both x buffers (pad columns 14..31 stay zero forever)
        for (int idx = tid; idx < 2 * TILE; idx += 256) s_in[idx] = 0;
    }

    // ---- prologue: tile 0 into buffer 0 ----
    if (LAYER == 0) {
        if (tid < 16 * I_SZ)
            s_in[xm * 32 + xk] = f2bf(xin[((size_t)(row0 + xm) * T_SZ + 0) * I_SZ + xk]);
    } else {
#ifdef HAVE_ASYNC_LDS
        wait_async0();
#else
        *reinterpret_cast<uint4*>(s_in + hm * H_SZ + hc8) =
            *reinterpret_cast<const uint4*>(hin + ((size_t)(row0 + hm) * T_SZ + 0) * H_SZ + hc8);
#endif
    }
    __syncthreads();

    // per-lane gate bias (depends only on the lane's output column);
    // added in the elementwise phase so accumulators can start from inline 0.
    float bias[4];
#pragma unroll
    for (int g = 0; g < 4; ++g) {
        int n = g * H_SZ + wave * 16 + col;
        bias[g] = b_ih[n] + b_hh[n];
    }

    v8f cstate;
#pragma unroll
    for (int v = 0; v < 8; ++v) cstate[v] = 0.0f;

    for (int t = 0; t < T_SZ; ++t) {
        // buffer selection by OFFSET (keeps LDS addrspace inference -> ds_load, not flat)
        const int curoff = (t & 1) * TILE;
        const int nxtoff = TILE - curoff;

        // ---- prefetch next input tile (overlaps with WMMA compute below) ----
        float xnext = 0.0f;
        if (LAYER == 0) {
            if (t + 1 < T_SZ && tid < 16 * I_SZ)
                xnext = xin[((size_t)(row0 + xm) * T_SZ + (t + 1)) * I_SZ + xk];
        } else {
            if (t + 1 < T_SZ) {
#ifdef HAVE_ASYNC_LDS
                async_copy16(hin + ((size_t)(row0 + hm) * T_SZ + (t + 1)) * H_SZ + hc8,
                             s_in + nxtoff + hm * H_SZ + hc8);
#else
                *reinterpret_cast<uint4*>(s_in + nxtoff + hm * H_SZ + hc8) =
                    *reinterpret_cast<const uint4*>(
                        hin + ((size_t)(row0 + hm) * T_SZ + (t + 1)) * H_SZ + hc8);
#endif
            }
        }

        v8f acc[4];
#pragma unroll
        for (int g = 0; g < 4; ++g)
#pragma unroll
            for (int v = 0; v < 8; ++v) acc[g][v] = 0.0f;   // -> inline-0 C on 1st WMMA

        // ---- input projection: gates += x_t @ w_ih^T ----
#pragma unroll
        for (int kk = 0; kk < KIN / 32; ++kk) {
            v16bf a = load_frag(s_in + curoff + col * KIN + kk * 32 + half * 8);
            v16bf bb[4];
#pragma unroll
            for (int g = 0; g < 4; ++g)
                bb[g] = load_frag(s_wih + (g * H_SZ + wave * 16 + col) * KIN + kk * 32 + half * 8);
#pragma unroll
            for (int g = 0; g < 4; ++g)
                acc[g] = __builtin_amdgcn_wmma_f32_16x16x32_bf16(
                    false, a, false, bb[g], (short)0, acc[g], false, false);
        }
        // ---- recurrent: gates += h_{t-1} @ w_hh^T ----
#pragma unroll
        for (int kk = 0; kk < H_SZ / 32; ++kk) {
            v16bf a = load_frag(s_h + col * H_SZ + kk * 32 + half * 8);
            v16bf bb[4];
#pragma unroll
            for (int g = 0; g < 4; ++g)
                bb[g] = load_frag(s_whh + (g * H_SZ + wave * 16 + col) * H_SZ + kk * 32 + half * 8);
#pragma unroll
            for (int g = 0; g < 4; ++g)
                acc[g] = __builtin_amdgcn_wmma_f32_16x16x32_bf16(
                    false, a, false, bb[g], (short)0, acc[g], false, false);
        }

        // ---- elementwise LSTM cell update (per-lane: 8 rows of one column) ----
        float hval[8];
#pragma unroll
        for (int v = 0; v < 8; ++v) {
            float iv = sigmoidf_(acc[0][v] + bias[0]);
            float fv = sigmoidf_(acc[1][v] + bias[1]);
            float gv = tanhf_(acc[2][v] + bias[2]);
            float ov = sigmoidf_(acc[3][v] + bias[3]);
            float cv = fv * cstate[v] + iv * gv;
            cstate[v] = cv;
            hval[v] = ov * tanhf_(cv);
        }

        __syncthreads();   // all fragment reads of s_h / current tile complete (WAR)

#pragma unroll
        for (int v = 0; v < 8; ++v)
            s_h[(half * 8 + v) * H_SZ + wave * 16 + col] = f2bf(hval[v]);

        if (LAYER == 0) {
            // commit prefetched x(t+1) into the back buffer (global latency hidden)
            if (t + 1 < T_SZ && tid < 16 * I_SZ)
                s_in[nxtoff + xm * 32 + xk] = f2bf(xnext);
            __syncthreads();  // h tile + next input buffer ready
            // spill h tile to global h1 sequence (coalesced 16B per thread)
            *reinterpret_cast<uint4*>(hout + ((size_t)(row0 + hm) * T_SZ + t) * H_SZ + hc8) =
                *reinterpret_cast<const uint4*>(s_h + hm * H_SZ + hc8);
        } else {
#ifdef HAVE_ASYNC_LDS
            wait_async0();    // next tile landed in LDS (no-op on last step)
#endif
            __syncthreads();  // h tile + next input buffer ready
        }
    }

    if (LAYER == 1) {
        // final linear: out[b] = h_last[b] . w_lin + b_lin  (16 rows per block)
        if (tid < 16) {
            float s = b_lin[0];
            for (int k = 0; k < H_SZ; ++k)
                s += bf2f(s_h[tid * H_SZ + k]) * w_lin[k];
            out[row0 + tid] = s;   // out shape [B, 1]
        }
    }
}

extern "C" void kernel_launch(void* const* d_in, const int* in_sizes, int n_in,
                              void* d_out, int out_size, void* d_ws, size_t ws_size,
                              hipStream_t stream) {
    const float* x     = (const float*)d_in[0];
    const float* w_ih0 = (const float*)d_in[1];
    const float* w_hh0 = (const float*)d_in[2];
    const float* b_ih0 = (const float*)d_in[3];
    const float* b_hh0 = (const float*)d_in[4];
    const float* w_ih1 = (const float*)d_in[5];
    const float* w_hh1 = (const float*)d_in[6];
    const float* b_ih1 = (const float*)d_in[7];
    const float* b_hh1 = (const float*)d_in[8];
    const float* w_lin = (const float*)d_in[9];
    const float* b_lin = (const float*)d_in[10];
    float* out = (float*)d_out;

    // workspace: layer-0 hidden sequence in bf16: 1024*256*128*2 = 64 MiB
    unsigned short* h1 = (unsigned short*)d_ws;

    const size_t lds0 = (size_t)(G4 * 32   + G4 * H_SZ + 16 * H_SZ + 2 * 16 * 32  ) * sizeof(unsigned short);
    const size_t lds1 = (size_t)(G4 * H_SZ + G4 * H_SZ + 16 * H_SZ + 2 * 16 * H_SZ) * sizeof(unsigned short);

    hipLaunchKernelGGL((lstm_layer_kernel<0>), dim3(B_SZ / 16), dim3(256), lds0, stream,
                       x, (const unsigned short*)nullptr, w_ih0, w_hh0, b_ih0, b_hh0,
                       h1, w_lin, b_lin, out);
    hipLaunchKernelGGL((lstm_layer_kernel<1>), dim3(B_SZ / 16), dim3(256), lds1, stream,
                       (const float*)nullptr, (const unsigned short*)h1, w_ih1, w_hh1,
                       b_ih1, b_hh1, (unsigned short*)nullptr, w_lin, b_lin, out);
}
